// Attention_27934467293614
// MI455X (gfx1250) — compile-verified, matrix operand order checked by reading, not run on hardware
//
#include <hip/hip_runtime.h>

typedef __attribute__((ext_vector_type(16))) _Float16 v16h;
typedef __attribute__((ext_vector_type(2)))  _Float16 v2h;
typedef __attribute__((ext_vector_type(8)))  float    v8f;
typedef __attribute__((ext_vector_type(8)))  int      v8i;

#define WAVES_PER_BLOCK 8
#define NB 16            // batches per wave
#define S_TOK 10
#define D_FEAT 6

// pack two f32 -> one dword of two f16 (single v_cvt_pk_rtz_f16_f32)
static __device__ __forceinline__ int pkf16(float a, float b) {
#if __has_builtin(__builtin_amdgcn_cvt_pkrtz)
    return __builtin_bit_cast(int, __builtin_amdgcn_cvt_pkrtz(a, b));
#else
    union { _Float16 h[2]; int i; } u;
    u.h[0] = (_Float16)a; u.h[1] = (_Float16)b;
    return u.i;
#endif
}

// swap the two 16-lane halves of a wave32 (pure VALU, no LDS)
static __device__ __forceinline__ int xhalf16(int v) {
#if __has_builtin(__builtin_amdgcn_permlanex16)
    return __builtin_amdgcn_permlanex16(v, v, 0x76543210, 0xfedcba98, false, false);
#else
    return __shfl_xor(v, 16, 32);
#endif
}
static __device__ __forceinline__ float xhalf16f(float v) {
    return __builtin_bit_cast(float, xhalf16(__builtin_bit_cast(int, v)));
}

// ---------------------------------------------------------------------------
// One-block prep kernel: pack fused projection weights into d_ws.
//   ws[0..255]  : 16 rows x 16B  f16 W'[m][kk0..7] (row m = out-feature)
//                 m 0-5 = Wq row m * log2(e), 6-11 = Wk row m-6, 12 = sum_e Wv
//   ws[256..319]: 16 f32 biases (bq*log2e, bk, sum bv)
// ---------------------------------------------------------------------------
__global__ void attn_prep_kernel(const float* __restrict__ Wk, const float* __restrict__ bk,
                                 const float* __restrict__ Wq, const float* __restrict__ bq,
                                 const float* __restrict__ Wv, const float* __restrict__ bv,
                                 float* __restrict__ ws)
{
    const int t = threadIdx.x;
    if (t >= 16) return;
    const float LOG2E = 1.4426950408889634f;
    _Float16 wrow[8];
    #pragma unroll
    for (int kk = 0; kk < 8; ++kk) wrow[kk] = (_Float16)0.0f;
    float bias = 0.0f;
    if (t < 6) {
        for (int kk = 0; kk < 6; ++kk) wrow[kk] = (_Float16)(Wq[t * 6 + kk] * LOG2E);
        bias = bq[t] * LOG2E;
    } else if (t < 12) {
        for (int kk = 0; kk < 6; ++kk) wrow[kk] = (_Float16)Wk[(t - 6) * 6 + kk];
        bias = bk[t - 6];
    } else if (t == 12) {
        for (int kk = 0; kk < 6; ++kk) {
            float s = 0.0f;
            for (int e = 0; e < 6; ++e) s += Wv[e * 6 + kk];
            wrow[kk] = (_Float16)s;
        }
        float s = 0.0f;
        for (int e = 0; e < 6; ++e) s += bv[e];
        bias = s;
    }
    ((float4*)ws)[t] = *(const float4*)wrow;   // 16B packed f16 row
    ws[64 + t] = bias;
}

// ---------------------------------------------------------------------------
// Main kernel: per batch, projection WMMA (transposed: D = W'^T x X^T) puts
// all 13 features of token n into lane n's accum regs (+ opposite half), then
// a scores^T WMMA (A=keys, B=queries) and a 2-permlane softmax.
// ---------------------------------------------------------------------------
__global__ __launch_bounds__(256) void attn6x10_wmma_kernel(
    const float* __restrict__ x, const float* __restrict__ ws,
    float* __restrict__ out, int Btotal)
{
    __shared__ float vsarr[WAVES_PER_BLOCK][NB][32];
    __shared__ float oscr [WAVES_PER_BLOCK][NB][32];

    const int lane = threadIdx.x & 31;
    const int wave = threadIdx.x >> 5;
    const int n    = lane & 15;
    const int hi   = lane >> 4;

    const long gwave     = (long)blockIdx.x * WAVES_PER_BLOCK + wave;
    const long batchBase = gwave * NB;
    if (batchBase >= Btotal) return;                     // wave-uniform
    const long rowBase   = batchBase * S_TOK;
    const long totalRows = (long)Btotal * S_TOK;

    // --- A operand: packed weight rows (out-feature per lane), from d_ws ---
    v8i bwi;
    #pragma unroll
    for (int i = 0; i < 8; ++i) bwi[i] = 0;
    if (hi == 0) {
        int4 w = ((const int4*)ws)[n];
        bwi[0] = w.x; bwi[1] = w.y; bwi[2] = w.z;        // 6 f16 live
    }
    const v16h bw = __builtin_bit_cast(v16h, bwi);

    // --- C operand for projection: bias[m], m = vi + 8*hi ---
    const float4* bb = (const float4*)(ws + 64 + 8 * hi);
    float4 b0 = bb[0], b1 = bb[1];
    v8f cb;
    cb[0]=b0.x; cb[1]=b0.y; cb[2]=b0.z; cb[3]=b0.w;
    cb[4]=b1.x; cb[5]=b1.y; cb[6]=b1.z; cb[7]=b1.w;

    // --- C operand for scores: -100 on key rows >= 10 -> exp2 -> 0 ---
    v8f cz;
    #pragma unroll
    for (int vi = 0; vi < 8; ++vi) cz[vi] = (vi + 8 * hi >= S_TOK) ? -100.0f : 0.0f;

    // x base for this lane (clamped duplicate rows for inactive lanes)
    const int  lrow  = (n < S_TOK) ? n : (S_TOK - 1);
    const float* xlane = x + (rowBase + lrow) * D_FEAT;
    const bool act = (hi == 0);

    // persistent fragments: upper elements stay zero across iterations
    v8i xfi, afi, bfi;
    #pragma unroll
    for (int i = 0; i < 8; ++i) { xfi[i] = 0; afi[i] = 0; bfi[i] = 0; }

    auto body = [&](int b) {
        // load this lane's token row (24B, 8B aligned)
        const float2* xb = (const float2*)(xlane + b * (S_TOK * D_FEAT));
        float2 p0 = xb[0], p1 = xb[1], p2 = xb[2];
        xfi[0] = act ? pkf16(p0.x, p0.y) : 0;
        xfi[1] = act ? pkf16(p1.x, p1.y) : 0;
        xfi[2] = act ? pkf16(p2.x, p2.y) : 0;

        // projection: D[m=feature][n=token]; lane n: m = vi + 8*hi
        v8f d = __builtin_amdgcn_wmma_f32_16x16x32_f16(
            false, bw, false, __builtin_bit_cast(v16h, xfi), (short)0, cb, false, false);

        // half0 lane n: d0-5 = q0-5, d6-7 = k0-1 ; half1 lane n: d0-3 = k2-5, d4 = vs
        vsarr[wave][b][lane] = d[4];                     // vs lives in slots 16-31
        int pk01 = pkf16(d[0], d[1]);
        int pk23 = pkf16(d[2], d[3]);
        int pk45 = pkf16(d[4], d[5]);
        int pk67 = pkf16(d[6], d[7]);                    // == 0 on upper half (zero rows)
        // pre-mask sender so upper half receives zeros; lower half receives k2-5
        int s01 = xhalf16(hi ? pk01 : 0);
        int s23 = xhalf16(hi ? pk23 : 0);
        afi[0] = pk67;                                   // keys: k0k1 (0 on upper half)
        afi[1] = s01;                                    // k2k3
        afi[2] = s23;                                    // k4k5
        bfi[0] = hi ? 0 : pk01;                          // queries: q0q1,q2q3,q4q5
        bfi[1] = hi ? 0 : pk23;
        bfi[2] = hi ? 0 : pk45;

        // scores^T: sc[m=key][n=query], key m = vi + 8*hi
        v8f sc = __builtin_amdgcn_wmma_f32_16x16x32_f16(
            false, __builtin_bit_cast(v16h, afi),
            false, __builtin_bit_cast(v16h, bfi), (short)0, cz, false, false);

        // per-key v-rowsums (broadcast b128 loads)
        const float4* vp = (const float4*)&vsarr[wave][b][16 + 8 * hi];
        float4 va = vp[0], vb2 = vp[1];

        float e0 = __builtin_amdgcn_exp2f(sc[0]);
        float e1 = __builtin_amdgcn_exp2f(sc[1]);
        float e2 = __builtin_amdgcn_exp2f(sc[2]);
        float e3 = __builtin_amdgcn_exp2f(sc[3]);
        float e4 = __builtin_amdgcn_exp2f(sc[4]);
        float e5 = __builtin_amdgcn_exp2f(sc[5]);
        float e6 = __builtin_amdgcn_exp2f(sc[6]);
        float e7 = __builtin_amdgcn_exp2f(sc[7]);
        float se = ((e0 + e1) + (e2 + e3)) + ((e4 + e5) + (e6 + e7));
        float sv = (fmaf(e1, va.y,  e0 * va.x)  + fmaf(e3, va.w,  e2 * va.z)) +
                   (fmaf(e5, vb2.y, e4 * vb2.x) + fmaf(e7, vb2.w, e6 * vb2.z));
        se += xhalf16f(se);                              // combine key halves (VALU)
        sv += xhalf16f(sv);
        oscr[wave][b][lane] = __fdividef(sv, se);        // query n result (dup halves)
    };

    if (batchBase + NB <= (long)Btotal) {
        #pragma unroll 4                                 // 4 batch pipelines in flight
        for (int b = 0; b < NB; ++b) body(b);
    } else {
        const int nb = (int)(Btotal - batchBase);
        for (int b = 0; b < nb; ++b) body(b);
    }

    // drain: coalesced store of NB*10 results
    for (int i = lane; i < NB * S_TOK; i += 32) {
        const int bb2 = i / S_TOK;
        const int r   = i - bb2 * S_TOK;
        const long gr = rowBase + i;
        if (gr < totalRows) out[gr] = oscr[wave][bb2][r];
    }
}

extern "C" void kernel_launch(void* const* d_in, const int* in_sizes, int n_in,
                              void* d_out, int out_size, void* d_ws, size_t ws_size,
                              hipStream_t stream) {
    const float* x  = (const float*)d_in[0];
    const float* Wk = (const float*)d_in[1];
    const float* bk = (const float*)d_in[2];
    const float* Wq = (const float*)d_in[3];
    const float* bq = (const float*)d_in[4];
    const float* Wv = (const float*)d_in[5];
    const float* bv = (const float*)d_in[6];
    float* out = (float*)d_out;
    float* ws  = (float*)d_ws;

    const int Btotal = in_sizes[0] / (S_TOK * D_FEAT);

    attn_prep_kernel<<<1, 32, 0, stream>>>(Wk, bk, Wq, bq, Wv, bv, ws);

    const int batchesPerBlock = WAVES_PER_BLOCK * NB;    // 128
    const int blocks = (Btotal + batchesPerBlock - 1) / batchesPerBlock;
    attn6x10_wmma_kernel<<<blocks, WAVES_PER_BLOCK * 32, 0, stream>>>(x, ws, out, Btotal);
}
